// CustomSimplexMappingAttention_72593537237254
// MI455X (gfx1250) — compile-verified, matrix operand order checked by reading, not run on hardware
//
#include <hip/hip_runtime.h>
#include <hip/hip_bf16.h>

// Sparsemax causal attention for MI455X (gfx1250, wave32, WMMA, async-LDS copies).
// B=2, L=2048, D=1024, H=16, HD=64.
//
// Pipeline (bf16 WMMA f32-accum):
//   1) cvt fp32->bf16 for x, Wq, Wk, Wv, Wo
//   2) q = x Wq^T -> [B,H,L,HD]; k likewise; v = x Wv^T -> [B,H,HD,L]
//      via a double-buffered GEMM that stages 128x32 A / 64x32 B tiles into LDS
//      with GLOBAL_LOAD_ASYNC_TO_LDS_B128 (ASYNCcnt) and feeds WMMA from ds reads.
//   3) attention: scores via WMMA into a 16x2048 fp32 LDS tile (128KB of the
//      320KB/WGP), causal mask, sparsemax via tau-bisection, P@V via WMMA.
//   4) out = ao Wo^T -> fp32 d_out.

typedef __attribute__((ext_vector_type(16))) __bf16 v16bf;
typedef __attribute__((ext_vector_type(8)))  __bf16 v8bf;
typedef __attribute__((ext_vector_type(8)))  float  v8f;

#define B_  2
#define L_  2048
#define D_  1024
#define H_  16
#define HD_ 64
#define LSP 2064   // padded LDS row stride in floats for the attention tile

__device__ __forceinline__ v8f zero8() {
  v8f z;
#pragma unroll
  for (int i = 0; i < 8; ++i) z[i] = 0.f;
  return z;
}

__device__ __forceinline__ v8f wmma_bf16(v16bf a, v16bf b, v8f c) {
  // (neg_a, A, neg_b, B, c_mod, C, reuse_a, reuse_b)
  return __builtin_amdgcn_wmma_f32_16x16x32_bf16(false, a, false, b, (short)0, c,
                                                 false, false);
}

// ---- CDNA5 async copy: global memory -> LDS, 16 bytes per lane -------------
// GV mode: vdst = wave-relative LDS byte address (low 32 bits of the generic
// shared pointer), vaddr = 64-bit global address. Tracked by ASYNCcnt.
__device__ __forceinline__ void async_cp16(unsigned lds_off, const void* g) {
  asm volatile("global_load_async_to_lds_b128 %0, %1, off"
               :: "v"(lds_off), "v"((unsigned long long)(uintptr_t)g)
               : "memory");
}
__device__ __forceinline__ void wait_async0() {
  asm volatile("s_wait_asynccnt 0x0" ::: "memory");
}

// WMMA 16x16x32 bf16 operand fragment from a row-major [rows, ld] matrix,
// contraction along ld (K). lane&15 selects the M/N row, lane>>4 the K
// half-group; per lane: two contiguous 16-byte loads at k0+8*hi and k0+16+8*hi.
__device__ __forceinline__ v16bf load_frag_nt(const __bf16* base, int ld,
                                              int row, int k0, int hi) {
  const __bf16* p = base + (size_t)row * ld + k0 + 8 * hi;
  v8bf a0 = *(const v8bf*)(p);
  v8bf a1 = *(const v8bf*)(p + 16);
  v16bf r;
#pragma unroll
  for (int i = 0; i < 8; ++i) { r[i] = a0[i]; r[i + 8] = a1[i]; }
  return r;
}

// Same fragment gather from an LDS tile with row stride 32 elements (64 B).
__device__ __forceinline__ v16bf load_frag_lds(const __bf16* tile, int row, int hi) {
  const __bf16* p = tile + row * 32 + 8 * hi;
  v8bf a0 = *(const v8bf*)(p);
  v8bf a1 = *(const v8bf*)(p + 16);
  v16bf r;
#pragma unroll
  for (int i = 0; i < 8; ++i) { r[i] = a0[i]; r[i + 8] = a1[i]; }
  return r;
}

__device__ __forceinline__ float wave_max32(float v) {
#pragma unroll
  for (int m = 16; m > 0; m >>= 1) v = fmaxf(v, __shfl_xor(v, m, 32));
  return v;
}
__device__ __forceinline__ float wave_sum32(float v) {
#pragma unroll
  for (int m = 16; m > 0; m >>= 1) v += __shfl_xor(v, m, 32);
  return v;
}

__global__ void cvt_f32_to_bf16(const float* __restrict__ src,
                                __bf16* __restrict__ dst, int n) {
  int i = blockIdx.x * blockDim.x + threadIdx.x;
  if (i < n) dst[i] = (__bf16)src[i];
}

// C[M,N] = A[M,K] x B[N,K]^T, bf16 in, f32 accumulate.
// Block = 256 threads = 8 waves; block tile 128(M) x 64(N); K-step 32.
// Double-buffered async staging: A tile 128x32 (8KB), B tile 64x32 (4KB) per
// buffer -> 24KB static LDS. Wave w owns rows w*16..w*16+15 across all 64 N.
// mode 0: store bf16 head-split     [B,H,L,HD]
// mode 1: store bf16 head-transpose [B,H,HD,L]
// mode 3: store f32 row-major       [M,N]
__global__ __launch_bounds__(256)
void gemm_nt_bf16(const __bf16* __restrict__ A, const __bf16* __restrict__ Bm,
                  void* __restrict__ Cout, int M, int N, int K, int mode) {
  __shared__ __bf16 tA[2][128 * 32];
  __shared__ __bf16 tB[2][64 * 32];

  const int t    = threadIdx.x;
  const int lane = t & 31;
  const int wave = t >> 5;
  const int hi   = lane >> 4;
  const int ln   = lane & 15;
  const int blockM = blockIdx.y * 128;
  const int nBase  = blockIdx.x * 64;

  v8f acc[4];
#pragma unroll
  for (int q = 0; q < 4; ++q) acc[q] = zero8();

  // Per-thread async staging of one K-step: A 8KB (2 chunks), B 4KB (1 chunk).
  auto stage = [&](int buf, int k0) {
#pragma unroll
    for (int p = 0; p < 2; ++p) {
      const int li  = t + p * 256;
      const int row = li >> 2;          // 0..127
      const int co  = (li & 3) * 8;     // bf16 elements (16B chunks)
      async_cp16((unsigned)(uintptr_t)(&tA[buf][row * 32 + co]),
                 A + (size_t)(blockM + row) * K + k0 + co);
    }
    {
      const int row = t >> 2;           // 0..63
      const int co  = (t & 3) * 8;
      async_cp16((unsigned)(uintptr_t)(&tB[buf][row * 32 + co]),
                 Bm + (size_t)(nBase + row) * K + k0 + co);
    }
  };

  const int KT = K >> 5;                // K-steps of 32
  stage(0, 0);
  for (int kt = 0; kt < KT; ++kt) {
    const int buf = kt & 1;
    wait_async0();                      // this wave's async stores to LDS done
    __syncthreads();                    // every wave's share visible
    if (kt + 1 < KT) stage(buf ^ 1, (kt + 1) * 32);  // overlap DMA w/ compute

    v16bf a = load_frag_lds(tA[buf], wave * 16 + ln, hi);
#pragma unroll
    for (int q = 0; q < 4; ++q) {
      v16bf b = load_frag_lds(tB[buf], q * 16 + ln, hi);
      acc[q] = wmma_bf16(a, b, acc[q]);
    }
    __syncthreads();                    // all reads done before buf is re-staged
  }

  const int mBase = blockM + wave * 16;
#pragma unroll
  for (int q = 0; q < 4; ++q) {
    const int cg = nBase + q * 16 + ln;
#pragma unroll
    for (int i = 0; i < 8; ++i) {
      const int r = mBase + i + 8 * hi;       // global M row
      const float val = acc[q][i];
      if (mode == 0) {
        const int b = r >> 11, l = r & (L_ - 1);
        const int h = cg >> 6, hd = cg & (HD_ - 1);
        ((__bf16*)Cout)[(((size_t)(b * H_ + h) * L_ + l) << 6) + hd] = (__bf16)val;
      } else if (mode == 1) {
        const int b = r >> 11, l = r & (L_ - 1);
        const int h = cg >> 6, hd = cg & (HD_ - 1);
        ((__bf16*)Cout)[((size_t)(b * H_ + h) * HD_ + hd) * L_ + l] = (__bf16)val;
      } else {
        ((float*)Cout)[(size_t)r * N + cg] = val;
      }
    }
  }
}

// One block per (b, h, 16-query tile). 256 threads = 8 waves.
// q,k: [B,H,L,HD] bf16; vT: [B,H,HD,L] bf16; ao: [B*L, D] bf16.
__global__ __launch_bounds__(256)
void sparsemax_attn_kernel(const __bf16* __restrict__ q,
                           const __bf16* __restrict__ k,
                           const __bf16* __restrict__ vT,
                           __bf16* __restrict__ ao) {
  extern __shared__ float smem[];            // [16][LSP] score/prob tile
  const int lane = threadIdx.x & 31;
  const int wave = threadIdx.x >> 5;
  const int hi   = lane >> 4;
  const int ln   = lane & 15;

  const int qt = blockIdx.x & (L_ / 16 - 1); // query tile
  const int bh = blockIdx.x >> 7;            // b*H + h
  const int qBase = qt * 16;
  const int nkeys = qBase + 16;              // causal coverage for this tile
  const int nkc   = (nkeys + 31) >> 5;       // 32-key chunks
  const int span  = nkc * 32;

  const __bf16* qp = q  + (size_t)bh * L_ * HD_;
  const __bf16* kp = k  + (size_t)bh * L_ * HD_;
  const __bf16* vp = vT + (size_t)bh * HD_ * L_;

  // ---- Phase 1: scores S = (Q K^T)/8 with causal mask, into LDS ----
  for (int c = wave; c < nkc; c += 8) {      // wave-uniform trip count
#pragma unroll
    for (int half = 0; half < 2; ++half) {
      const int keyBase = c * 32 + half * 16;
      v8f acc = zero8();
#pragma unroll
      for (int kk = 0; kk < HD_; kk += 32) {
        v16bf a = load_frag_nt(qp, HD_, qBase + ln, kk, hi);
        v16bf b = load_frag_nt(kp, HD_, keyBase + ln, kk, hi);
        acc = wmma_bf16(a, b, acc);
      }
#pragma unroll
      for (int i = 0; i < 8; ++i) {
        const int rl  = i + 8 * hi;          // local query row 0..15
        const int key = keyBase + ln;
        float s = acc[i] * 0.125f;           // 1/sqrt(64)
        if (key > qBase + rl) s = -1e9f;     // causal NEG
        smem[rl * LSP + key] = s;
      }
    }
  }
  __syncthreads();

  // ---- Phase 2: sparsemax per row via bisection on tau ----
  // f(tau) = sum(max(z - tau, 0)) is monotone piecewise-linear; solve f(tau)=1
  // on [max-1, max] to fp32 precision (exact simplex projection).
#pragma unroll
  for (int rr = 0; rr < 2; ++rr) {
    const int rl = wave * 2 + rr;            // 8 waves x 2 rows = 16 rows
    const int n  = qBase + rl + 1;           // causal length
    float* row = smem + rl * LSP;
    float m = -3.4e38f;
    for (int j = lane; j < n; j += 32) m = fmaxf(m, row[j]);
    m = wave_max32(m);
    float lo = m - 1.0f, hib = m;
    for (int it = 0; it < 40; ++it) {
      const float mid = 0.5f * (lo + hib);
      float s = 0.f;
      for (int j = lane; j < n; j += 32) s += fmaxf(row[j] - mid, 0.f);
      s = wave_sum32(s);
      if (s >= 1.0f) lo = mid; else hib = mid;
    }
    const float tau = lo;
    // write probabilities over the whole chunked span (NEG entries -> 0)
    for (int j = lane; j < span; j += 32) row[j] = fmaxf(row[j] - tau, 0.f);
  }
  __syncthreads();

  // ---- Phase 3: O = P @ V via WMMA; waves 0..3 each own a 16-wide HD tile ----
  if (wave < 4) {                            // wave-uniform
    const int nB = wave * 16;
    v8f acc = zero8();
    for (int c = 0; c < nkc; ++c) {
      // A fragment (P, 16x32) gathered from LDS: two 32B contiguous runs/lane
      const float* rp = smem + ln * LSP + c * 32 + 8 * hi;
      float4 f0 = *(const float4*)(rp);
      float4 f1 = *(const float4*)(rp + 4);
      float4 f2 = *(const float4*)(rp + 16);
      float4 f3 = *(const float4*)(rp + 20);
      v16bf a;
      a[0]=(__bf16)f0.x; a[1]=(__bf16)f0.y; a[2]=(__bf16)f0.z; a[3]=(__bf16)f0.w;
      a[4]=(__bf16)f1.x; a[5]=(__bf16)f1.y; a[6]=(__bf16)f1.z; a[7]=(__bf16)f1.w;
      a[8]=(__bf16)f2.x; a[9]=(__bf16)f2.y; a[10]=(__bf16)f2.z; a[11]=(__bf16)f2.w;
      a[12]=(__bf16)f3.x; a[13]=(__bf16)f3.y; a[14]=(__bf16)f3.z; a[15]=(__bf16)f3.w;
      // B fragment from transposed V ([HD,L] row-major -> contiguous loads)
      v16bf b = load_frag_nt(vp, L_, nB + ln, c * 32, hi);
      acc = wmma_bf16(a, b, acc);
    }
    const int bq = bh >> 4;                  // batch
    const int h  = bh & (H_ - 1);
#pragma unroll
    for (int i = 0; i < 8; ++i) {
      const int qrow = qBase + i + 8 * hi;
      ao[((size_t)(bq * L_ + qrow) * D_) + h * HD_ + nB + ln] = (__bf16)acc[i];
    }
  }
}

extern "C" void kernel_launch(void* const* d_in, const int* in_sizes, int n_in,
                              void* d_out, int out_size, void* d_ws, size_t ws_size,
                              hipStream_t stream) {
  (void)in_sizes; (void)n_in; (void)out_size; (void)ws_size;
  const float* x  = (const float*)d_in[0];
  const float* Wq = (const float*)d_in[1];
  const float* Wk = (const float*)d_in[2];
  const float* Wv = (const float*)d_in[3];
  const float* Wo = (const float*)d_in[4];

  // Workspace layout (bytes); total = 48 MB.
  char* ws = (char*)d_ws;
  const size_t XB  = (size_t)B_ * L_ * D_ * 2;       // 8 MB bf16 activations
  const size_t WB  = (size_t)D_ * D_ * 2;            // 2 MB bf16 weight
  const size_t QKB = (size_t)B_ * H_ * L_ * HD_ * 2; // 8 MB
  __bf16* xb = (__bf16*)(ws);
  __bf16* wq = (__bf16*)(ws + XB);
  __bf16* wk = (__bf16*)(ws + XB + WB);
  __bf16* wv = (__bf16*)(ws + XB + 2 * WB);
  __bf16* wo = (__bf16*)(ws + XB + 3 * WB);
  __bf16* qd = (__bf16*)(ws + XB + 4 * WB);
  __bf16* kd = (__bf16*)(ws + XB + 4 * WB + QKB);
  __bf16* vT = (__bf16*)(ws + XB + 4 * WB + 2 * QKB);
  __bf16* ao = (__bf16*)(ws + XB + 4 * WB + 3 * QKB);

  const int nx = B_ * L_ * D_;   // 4,194,304
  const int nw = D_ * D_;        // 1,048,576
  cvt_f32_to_bf16<<<(nx + 255) / 256, 256, 0, stream>>>(x,  xb, nx);
  cvt_f32_to_bf16<<<(nw + 255) / 256, 256, 0, stream>>>(Wq, wq, nw);
  cvt_f32_to_bf16<<<(nw + 255) / 256, 256, 0, stream>>>(Wk, wk, nw);
  cvt_f32_to_bf16<<<(nw + 255) / 256, 256, 0, stream>>>(Wv, wv, nw);
  cvt_f32_to_bf16<<<(nw + 255) / 256, 256, 0, stream>>>(Wo, wo, nw);

  const int M = B_ * L_;                       // 4096
  dim3 ggrid(D_ / 64, M / 128);                // (16, 32) blocks of 128x64
  gemm_nt_bf16<<<ggrid, 256, 0, stream>>>(xb, wq, qd, M, D_, D_, 0);
  gemm_nt_bf16<<<ggrid, 256, 0, stream>>>(xb, wk, kd, M, D_, D_, 0);
  gemm_nt_bf16<<<ggrid, 256, 0, stream>>>(xb, wv, vT, M, D_, D_, 1);

  const int nblk = B_ * H_ * (L_ / 16);        // 4096
  const size_t lds = 16 * LSP * sizeof(float); // 132,096 B (fits 320KB/WGP)
  sparsemax_attn_kernel<<<nblk, 256, lds, stream>>>(qd, kd, vT, ao);

  gemm_nt_bf16<<<ggrid, 256, 0, stream>>>(ao, wo, d_out, M, D_, D_, 3);
}
